// SparseVisionAttention_57672820850760
// MI455X (gfx1250) — matrix operand order repaired and checked by reading, not compiled, not run on hardware
//
#include <hip/hip_runtime.h>
#include <hip/hip_bf16.h>
#include <math.h>

// ---------------- problem constants ----------------
#define B_   8
#define N_   1024
#define E_   768
#define H_   12
#define D_   64
#define NG_  16
#define NT_  1040              // N_ + NG_
#define NTP_ 1056              // NT_ padded to multiple of 32 (vT rows)
#define NTILES_ 65             // NT_ / 16
#define JT4_ 17                // ceil(NTILES_ / 4)
#define MTILES_QKV ((B_ * NT_) / 16)   // 520
#define MTILES_OUT ((B_ * N_) / 16)    // 512
#define SCALE_Q 0.125f         // D^-0.5

typedef __attribute__((ext_vector_type(16))) _Float16 v16h;
typedef __attribute__((ext_vector_type(8)))  _Float16 h8;
typedef __attribute__((ext_vector_type(8)))  float    v8f;
typedef __attribute__((ext_vector_type(4)))  float    f4;

static __device__ inline v8f wmma_f16(v16h a, v16h b, v8f c) {
  return __builtin_amdgcn_wmma_f32_16x16x32_f16(false, a, false, b, (short)0, c, false, false);
}

static __device__ inline v16h make_v16h(h8 lo, h8 hi) {
  v16h r;
#pragma unroll
  for (int i = 0; i < 8; ++i) { r[i] = lo[i]; r[8 + i] = hi[i]; }
  return r;
}

// A fragment, f16, 16x32 (MxK), row-major source, K contiguous.
// lane: M = lane&15; kb = (lane<16)?0:8; halves = K{kb..kb+7}, K{16+kb..16+kb+7}
static __device__ inline v16h load_a_f16(const _Float16* __restrict__ A, int lda, int lane) {
  const _Float16* p = A + (lane & 15) * lda + ((lane & 16) ? 8 : 0);
  h8 lo = *reinterpret_cast<const h8*>(p);
  h8 hi = *reinterpret_cast<const h8*>(p + 16);
  return make_v16h(lo, hi);
}

// B fragment, f16, 32x16 (KxN), per-lane column n has K contiguous in memory.
// lane: N = lane&15; kb = (lane<16)?0:8; halves = K{kb..kb+7}, K{16+kb..16+kb+7}
static __device__ inline v16h load_b_kcont(const _Float16* __restrict__ Bm, int ldb, int lane) {
  const _Float16* p = Bm + (long)(lane & 15) * ldb + ((lane & 16) ? 8 : 0);
  h8 lo = *reinterpret_cast<const h8*>(p);
  h8 hi = *reinterpret_cast<const h8*>(p + 16);
  return make_v16h(lo, hi);
}

// A fragment from fp32 source (attention probs), converted to f16 in registers.
// kvalid = 32 or 16 (never read past row element kvalid).
static __device__ inline v16h load_a_attn(const float* __restrict__ A, int lda,
                                          int lane, int kvalid) {
  const float* p = A + (long)(lane & 15) * lda + ((lane & 16) ? 8 : 0);
  f4 a0 = *reinterpret_cast<const f4*>(p);
  f4 a1 = *reinterpret_cast<const f4*>(p + 4);
  v16h r;
#pragma unroll
  for (int i = 0; i < 4; ++i) { r[i] = (_Float16)a0[i]; r[4 + i] = (_Float16)a1[i]; }
  if (kvalid == 32) {
    f4 a2 = *reinterpret_cast<const f4*>(p + 16);
    f4 a3 = *reinterpret_cast<const f4*>(p + 20);
#pragma unroll
    for (int i = 0; i < 4; ++i) { r[8 + i] = (_Float16)a2[i]; r[12 + i] = (_Float16)a3[i]; }
  } else {
#pragma unroll
    for (int i = 0; i < 8; ++i) r[8 + i] = (_Float16)0.f;
  }
  return r;
}

static __device__ inline int iabs(int v) { return v < 0 ? -v : v; }

// ---------------- kernel 1: build xg (concat global tokens + x) as f16 ----------------
__global__ void __launch_bounds__(256)
build_xg_kernel(const float* __restrict__ x, const float* __restrict__ g,
                _Float16* __restrict__ xg) {
  long i = (long)blockIdx.x * blockDim.x + threadIdx.x;
  const long total = (long)B_ * NT_ * E_;
  if (i >= total) return;
  int c = (int)(i % E_);
  long bn = i / E_;
  int tok = (int)(bn % NT_);
  int b = (int)(bn / NT_);
  float val = (tok < NG_) ? g[tok * E_ + c]
                          : x[((long)b * N_ + (tok - NG_)) * E_ + c];
  xg[i] = (_Float16)val;
}

__global__ void __launch_bounds__(256)
cvt_f32_f16_kernel(const float* __restrict__ s, _Float16* __restrict__ d, int n) {
  int i = blockIdx.x * blockDim.x + threadIdx.x;
  if (i < n) d[i] = (_Float16)s[i];
}

// zero the 16-column pad of vT rows (NT_..NTP_-1), B_*H_*D_ rows
__global__ void __launch_bounds__(256)
zero_vt_pad_kernel(_Float16* __restrict__ vt) {
  int i = blockIdx.x * blockDim.x + threadIdx.x;
  const int total = B_ * H_ * D_ * (NTP_ - NT_);
  if (i >= total) return;
  int c = i & 15;                // NTP_-NT_ == 16
  int row = i >> 4;
  vt[(long)row * NTP_ + NT_ + c] = (_Float16)0.f;
}

// ---------------- kernel 2: fused QKV projection ----------------
// One wave per (16-row M tile) x (head). A frag reused 12x per K-step.
// q,k stored (B,H,Nt,D); v stored transposed (B,H,D,NTP_) for fast AV B-fragments.
__global__ void __launch_bounds__(256)
qkv_proj_kernel(const _Float16* __restrict__ xg,
                const _Float16* __restrict__ wq, const _Float16* __restrict__ wk,
                const _Float16* __restrict__ wv,
                const float* __restrict__ bq, const float* __restrict__ bk,
                const float* __restrict__ bv,
                _Float16* __restrict__ q, _Float16* __restrict__ k,
                _Float16* __restrict__ vt) {
  int wid = (int)(((long)blockIdx.x * blockDim.x + threadIdx.x) >> 5);
  int lane = threadIdx.x & 31;
  if (wid >= MTILES_QKV * H_) return;
  int mt = wid / H_;
  int h  = wid % H_;
  const _Float16* Arow = xg + (long)mt * 16 * E_;
  const v8f zero = {0.f, 0.f, 0.f, 0.f, 0.f, 0.f, 0.f, 0.f};
  v8f aq[4], ak[4], av4[4];
#pragma unroll
  for (int s = 0; s < 4; ++s) { aq[s] = zero; ak[s] = zero; av4[s] = zero; }
  for (int k0 = 0; k0 < E_; k0 += 32) {
    v16h a = load_a_f16(Arow + k0, E_, lane);
#pragma unroll
    for (int s = 0; s < 4; ++s) {
      long n0 = (long)(h * 64 + s * 16);
      v16h bw = load_b_kcont(wq + n0 * E_ + k0, E_, lane);
      aq[s] = wmma_f16(a, bw, aq[s]);
      bw = load_b_kcont(wk + n0 * E_ + k0, E_, lane);
      ak[s] = wmma_f16(a, bw, ak[s]);
      bw = load_b_kcont(wv + n0 * E_ + k0, E_, lane);
      av4[s] = wmma_f16(a, bw, av4[s]);
    }
  }
  int nl = lane & 15;
  int mo = (lane & 16) ? 8 : 0;
  int row0 = mt * 16;
  int b = row0 / NT_;            // Nt % 16 == 0 -> tile never crosses batch
  int tok = row0 % NT_;
  long bh = (long)b * H_ + h;
  long base = (bh * NT_ + tok + mo) * D_;
  long vtbase = bh * D_ * NTP_;
#pragma unroll
  for (int s = 0; s < 4; ++s) {
    int d = s * 16 + nl;
    float fbq = bq[h * 64 + d], fbk = bk[h * 64 + d], fbv = bv[h * 64 + d];
    h8 vv;
#pragma unroll
    for (int r = 0; r < 8; ++r) {
      q[base + (long)r * D_ + d] = (_Float16)((aq[s][r] + fbq) * SCALE_Q);
      k[base + (long)r * D_ + d] = (_Float16)(ak[s][r] + fbk);
      vv[r] = (_Float16)(av4[s][r] + fbv);
    }
    // vT: lane owns fixed d, 8 consecutive tokens -> one 16B store
    *reinterpret_cast<h8*>(vt + vtbase + (long)d * NTP_ + tok + mo) = vv;
  }
}

// ---------------- kernel 3: masked scores = Q x K^T (unnormalized) ----------------
// One wave per 16x64 strip (4 j-tiles): Q fragment reused 4x per K-step.
__global__ void __launch_bounds__(256)
scores_kernel(const _Float16* __restrict__ q, const _Float16* __restrict__ k,
              float* __restrict__ attn) {
  long wid = ((long)blockIdx.x * blockDim.x + threadIdx.x) >> 5;
  int lane = threadIdx.x & 31;
  if (wid >= (long)B_ * H_ * NTILES_ * JT4_) return;
  int j4 = (int)(wid % JT4_);
  long t = wid / JT4_;
  int it = (int)(t % NTILES_);
  int bh = (int)(t / NTILES_);
  const _Float16* Q = q + ((long)bh * NT_ + it * 16) * D_;
  const _Float16* K = k + (long)bh * NT_ * D_;
  const v8f zero = {0.f, 0.f, 0.f, 0.f, 0.f, 0.f, 0.f, 0.f};
  v8f acc[4];
#pragma unroll
  for (int s = 0; s < 4; ++s) acc[s] = zero;
#pragma unroll
  for (int k0 = 0; k0 < D_; k0 += 32) {
    v16h a = load_a_f16(Q + k0, D_, lane);
#pragma unroll
    for (int s = 0; s < 4; ++s) {
      int jt = j4 * 4 + s;
      if (jt < NTILES_) {
        v16h bw = load_b_kcont(K + (long)jt * 16 * D_ + k0, D_, lane);
        acc[s] = wmma_f16(a, bw, acc[s]);
      }
    }
  }
  int i0 = it * 16 + ((lane & 16) ? 8 : 0);
#pragma unroll
  for (int s = 0; s < 4; ++s) {
    int jt = j4 * 4 + s;
    if (jt >= NTILES_) continue;
    int j = jt * 16 + (lane & 15);
    float* o = attn + ((long)bh * NT_ + i0) * NT_ + j;
    int rj = (j - NG_) >> 5, cj = (j - NG_) & 31;   // s_grid = ceil(sqrt(1024)) = 32
#pragma unroll
    for (int r = 0; r < 8; ++r) {
      int i = i0 + r;
      bool vis = (i < NG_) || (j < NG_);
      if (!vis) {
        int ri = (i - NG_) >> 5, ci = (i - NG_) & 31;
        vis = (iabs(ri - rj) <= 3) && (iabs(ci - cj) <= 3);   // window 7 -> half 3
      }
      o[(long)r * NT_] = vis ? acc[s][r] : -INFINITY;
    }
  }
}

// ---------------- kernel 4: row softmax in place in d_out attn region ----------------
__global__ void __launch_bounds__(256)
softmax_kernel(float* __restrict__ attn) {
  long row = blockIdx.x;                    // B*H*NT rows
  float* p = attn + row * NT_;
  int t = threadIdx.x;
  float vals[5];
  float m = -INFINITY;
#pragma unroll
  for (int i = 0; i < 5; ++i) {
    int idx = t + i * 256;
    vals[i] = (idx < NT_) ? p[idx] : -INFINITY;
    m = fmaxf(m, vals[i]);
  }
#pragma unroll
  for (int off = 16; off >= 1; off >>= 1) m = fmaxf(m, __shfl_xor(m, off, 32));
  __shared__ float red[8];
  int w = t >> 5, lane = t & 31;
  if (lane == 0) red[w] = m;
  __syncthreads();
  m = red[0];
#pragma unroll
  for (int i = 1; i < 8; ++i) m = fmaxf(m, red[i]);
  __syncthreads();
  float s = 0.f;
#pragma unroll
  for (int i = 0; i < 5; ++i) {
    float e = __expf(vals[i] - m);          // exp(-inf - m) == 0 for masked / padded
    vals[i] = e;
    s += e;
  }
#pragma unroll
  for (int off = 16; off >= 1; off >>= 1) s += __shfl_xor(s, off, 32);
  if (lane == 0) red[w] = s;
  __syncthreads();
  s = 0.f;
#pragma unroll
  for (int i = 0; i < 8; ++i) s += red[i];
  float inv = 1.f / s;
#pragma unroll
  for (int i = 0; i < 5; ++i) {
    int idx = t + i * 256;
    if (idx < NT_) p[idx] = vals[i] * inv;
  }
}

// ---------------- kernel 5: out_h = attn x V  (f16 into (B,Nt,E) layout) ----------------
// V consumed from transposed vT (B,H,D,NTP_): B-fragment = two contiguous b128 loads.
__global__ void __launch_bounds__(256)
av_kernel(const float* __restrict__ attn, const _Float16* __restrict__ vt,
          _Float16* __restrict__ oh) {
  int wid = (int)(((long)blockIdx.x * blockDim.x + threadIdx.x) >> 5);
  int lane = threadIdx.x & 31;
  if (wid >= B_ * H_ * NTILES_) return;
  int it = wid % NTILES_;
  int bh = wid / NTILES_;
  const float* A = attn + ((long)bh * NT_ + it * 16) * NT_;
  const _Float16* V = vt + (long)bh * D_ * NTP_;
  const v8f zero = {0.f, 0.f, 0.f, 0.f, 0.f, 0.f, 0.f, 0.f};
  v8f acc[4];
#pragma unroll
  for (int s = 0; s < 4; ++s) acc[s] = zero;
  for (int j0 = 0; j0 < NT_; j0 += 32) {
    int kv = (j0 + 32 <= NT_) ? 32 : 16;    // 1040 = 32*32 + 16
    if (j0 + 32 < NT_)
      __builtin_prefetch((const void*)(A + (long)(lane & 15) * NT_ + j0 + 32), 0, 0);
    v16h a = load_a_attn(A + j0, NT_, lane, kv);
#pragma unroll
    for (int s = 0; s < 4; ++s) {
      // column d = s*16 + (lane&15); j contiguous; pad columns are zeroed
      v16h bf = load_b_kcont(V + (long)(s * 16) * NTP_ + j0, NTP_, lane);
      acc[s] = wmma_f16(a, bf, acc[s]);
    }
  }
  int b = bh / H_, h = bh % H_;
  int nl = lane & 15, mo = (lane & 16) ? 8 : 0;
  _Float16* o = oh + ((long)b * NT_ + it * 16 + mo) * E_ + h * 64;
#pragma unroll
  for (int s = 0; s < 4; ++s)
#pragma unroll
    for (int r = 0; r < 8; ++r)
      o[(long)r * E_ + s * 16 + nl] = (_Float16)acc[s][r];
}

// ---------------- kernel 6: final projection, sliced to tokens >= NG ----------------
__global__ void __launch_bounds__(256)
out_proj_kernel(const _Float16* __restrict__ oh, const _Float16* __restrict__ wo,
                const float* __restrict__ bo, float* __restrict__ out) {
  int wid = (int)(((long)blockIdx.x * blockDim.x + threadIdx.x) >> 5);
  int lane = threadIdx.x & 31;
  if (wid >= MTILES_OUT * H_) return;
  int mt = wid / H_;
  int h  = wid % H_;
  int b  = (mt * 16) / N_;
  int t0 = (mt * 16) % N_;
  const _Float16* Arow = oh + ((long)b * NT_ + NG_ + t0) * E_;   // skip global tokens
  const v8f zero = {0.f, 0.f, 0.f, 0.f, 0.f, 0.f, 0.f, 0.f};
  v8f acc[4];
#pragma unroll
  for (int s = 0; s < 4; ++s) acc[s] = zero;
  for (int k0 = 0; k0 < E_; k0 += 32) {
    v16h a = load_a_f16(Arow + k0, E_, lane);
#pragma unroll
    for (int s = 0; s < 4; ++s) {
      v16h bw = load_b_kcont(wo + (long)(h * 64 + s * 16) * E_ + k0, E_, lane);
      acc[s] = wmma_f16(a, bw, acc[s]);
    }
  }
  int nl = lane & 15, mo = (lane & 16) ? 8 : 0;
  float* o = out + ((long)b * N_ + t0 + mo) * E_ + h * 64;
#pragma unroll
  for (int s = 0; s < 4; ++s) {
    float fb = bo[h * 64 + s * 16 + nl];
#pragma unroll
    for (int r = 0; r < 8; ++r)
      o[(long)r * E_ + s * 16 + nl] = acc[s][r] + fb;
  }
}

// ---------------- host launcher ----------------
extern "C" void kernel_launch(void* const* d_in, const int* in_sizes, int n_in,
                              void* d_out, int out_size, void* d_ws, size_t ws_size,
                              hipStream_t stream) {
  const float* x  = (const float*)d_in[0];
  const float* g  = (const float*)d_in[1];
  const float* wq = (const float*)d_in[2];
  const float* bq = (const float*)d_in[3];
  const float* wk = (const float*)d_in[4];
  const float* bk = (const float*)d_in[5];
  const float* wv = (const float*)d_in[6];
  const float* bv = (const float*)d_in[7];
  const float* wo = (const float*)d_in[8];
  const float* bo = (const float*)d_in[9];

  float* out  = (float*)d_out;
  float* attn = out + (long)B_ * N_ * E_;     // d_out = [out | attn] flat

  // workspace layout (all 16B aligned)
  const size_t sz_xg = (size_t)B_ * NT_ * E_ * sizeof(_Float16);
  const size_t sz_w  = (size_t)E_ * E_ * sizeof(_Float16);
  const size_t sz_p  = (size_t)B_ * H_ * NT_ * D_ * sizeof(_Float16);
  const size_t sz_vt = (size_t)B_ * H_ * D_ * NTP_ * sizeof(_Float16);
  const size_t need  = sz_xg + 4 * sz_w + 2 * sz_p + sz_vt + sz_xg;
  if (ws_size < need) return;
  char* ws = (char*)d_ws;
  _Float16* xg  = (_Float16*)ws;  ws += sz_xg;
  _Float16* wqh = (_Float16*)ws;  ws += sz_w;
  _Float16* wkh = (_Float16*)ws;  ws += sz_w;
  _Float16* wvh = (_Float16*)ws;  ws += sz_w;
  _Float16* woh = (_Float16*)ws;  ws += sz_w;
  _Float16* qh  = (_Float16*)ws;  ws += sz_p;
  _Float16* kh  = (_Float16*)ws;  ws += sz_p;
  _Float16* vth = (_Float16*)ws;  ws += sz_vt;
  _Float16* oh  = (_Float16*)ws;  ws += sz_xg;

  // 1. prepare f16 operands
  {
    long total = (long)B_ * NT_ * E_;
    build_xg_kernel<<<(int)((total + 255) / 256), 256, 0, stream>>>(x, g, xg);
  }
  cvt_f32_f16_kernel<<<(E_ * E_ + 255) / 256, 256, 0, stream>>>(wq, wqh, E_ * E_);
  cvt_f32_f16_kernel<<<(E_ * E_ + 255) / 256, 256, 0, stream>>>(wk, wkh, E_ * E_);
  cvt_f32_f16_kernel<<<(E_ * E_ + 255) / 256, 256, 0, stream>>>(wv, wvh, E_ * E_);
  cvt_f32_f16_kernel<<<(E_ * E_ + 255) / 256, 256, 0, stream>>>(wo, woh, E_ * E_);
  {
    int total = B_ * H_ * D_ * (NTP_ - NT_);
    zero_vt_pad_kernel<<<(total + 255) / 256, 256, 0, stream>>>(vth);
  }

  // 2. fused QKV projection: 520 M-tiles x 12 heads waves
  {
    int waves = MTILES_QKV * H_;
    qkv_proj_kernel<<<(waves * 32 + 255) / 256, 256, 0, stream>>>(
        xg, wqh, wkh, wvh, bq, bk, bv, qh, kh, vth);
  }

  // 3. masked scores straight into d_out attn region (16x64 strips)
  {
    long waves = (long)B_ * H_ * NTILES_ * JT4_;      // 106080
    scores_kernel<<<(int)((waves * 32 + 255) / 256), 256, 0, stream>>>(qh, kh, attn);
  }

  // 4. softmax in place (one block per row)
  softmax_kernel<<<B_ * H_ * NT_, 256, 0, stream>>>(attn);

  // 5. attn x V (vT-fed B fragments)
  {
    int waves = B_ * H_ * NTILES_;                    // 6240
    av_kernel<<<(waves * 32 + 255) / 256, 256, 0, stream>>>(attn, vth, oh);
  }

  // 6. output projection (tokens >= NG only)
  {
    int waves = MTILES_OUT * H_;                      // 6144
    out_proj_kernel<<<(waves * 32 + 255) / 256, 256, 0, stream>>>(oh, woh, bo, out);
  }
}
